// LocalAttention_31104153158060
// MI455X (gfx1250) — matrix operand order repaired and checked by reading, not compiled, not run on hardware
//
#include <hip/hip_runtime.h>

// ---------------------------------------------------------------------------
// Problem constants (from reference): B=4, S=1024, EMBED=768, HEADS=12, D=64
// windows (16,64,256) nest and the mask is multiplicative -> effective band
// is |i-j|<=16 with out-of-band scores exactly 0 (NOT -inf).
// ---------------------------------------------------------------------------
#define BB   4
#define SS   1024
#define EE   768
#define HH   12
#define DD   64
#define MM   (BB * SS)   // 4096 rows for the projection GEMMs
#define WIN  16
#define BANDW (2 * WIN + 1)

typedef __attribute__((ext_vector_type(16))) __bf16 v16bf;
typedef __attribute__((ext_vector_type(8)))  float  v8f;

union Frag16 {
    uint4 q[2];   // 32 bytes = 16 bf16
    v16bf v;
};

// ---------------------------------------------------------------------------
// f32 -> bf16 (round-to-nearest-even) bulk convert
// ---------------------------------------------------------------------------
__global__ __launch_bounds__(256) void cvt_bf16_kernel(const float* __restrict__ in,
                                                       unsigned short* __restrict__ out,
                                                       int n) {
    int i = blockIdx.x * 256 + threadIdx.x;
    if (i < n) {
        unsigned u = __float_as_uint(in[i]);
        unsigned r = u + 0x7FFFu + ((u >> 16) & 1u);
        out[i] = (unsigned short)(r >> 16);
    }
}

// ---------------------------------------------------------------------------
// C[M,N] = A[M,K] * B[N,K]^T + bias[N]    (A,B in bf16 bits; C f32)
// Block tile 128x128, 8 waves, each wave: 64(M) x 32(N) = 4x2 WMMA tiles.
// Per-lane fragment layouts per CDNA5 ISA 7.12.2:
//   A (16x32 bf16): lane m / m+16 holds row (m&15); K runs
//     [k0+8*(lane>=16) .. +7] and [k0+8*(lane>=16)+16 .. +23]  -> 2x b128
//   B (32x16 bf16): lane n / n+16 holds col (n&15); K runs
//     [k0+8*(lane>=16) .. +7] and [k0+8*(lane>=16)+16 .. +23]  -> 2x b128
// Both runs are contiguous rows of X / W in memory (since C = X * W^T).
// ---------------------------------------------------------------------------
__global__ __launch_bounds__(256) void wmma_gemm_bias_kernel(
    const unsigned short* __restrict__ A,   // [M,K] bf16
    const unsigned short* __restrict__ Bm,  // [N,K] bf16 (W as given: [out,in])
    const float* __restrict__ bias,         // [N]
    float* __restrict__ C,                  // [M,N] f32
    int M, int N, int K)
{
    const int lane  = threadIdx.x & 31;
    const int wave  = threadIdx.x >> 5;
    const int waveM = wave & 1;   // 0..1
    const int waveN = wave >> 1;  // 0..3
    const int mb = blockIdx.y * 128 + waveM * 64;
    const int nb = blockIdx.x * 128 + waveN * 32;
    const int l15 = lane & 15;
    const int lhi = lane >> 4;    // 0 or 1

    v8f acc[4][2];
#pragma unroll
    for (int t = 0; t < 4; ++t)
#pragma unroll
        for (int u = 0; u < 2; ++u)
            acc[t][u] = (v8f)(0.0f);

    const int aKoff = lhi * 8;
    const int bKoff = lhi * 8;

    for (int k0 = 0; k0 < K; k0 += 32) {
        Frag16 a[4], b[2];
#pragma unroll
        for (int t = 0; t < 4; ++t) {
            const unsigned short* p =
                A + (size_t)(mb + t * 16 + l15) * K + (k0 + aKoff);
            a[t].q[0] = *(const uint4*)(p);        // K = k0+aKoff .. +7
            a[t].q[1] = *(const uint4*)(p + 16);   // K = k0+aKoff+16 .. +23
        }
#pragma unroll
        for (int u = 0; u < 2; ++u) {
            const unsigned short* p =
                Bm + (size_t)(nb + u * 16 + l15) * K + (k0 + bKoff);
            b[u].q[0] = *(const uint4*)(p);        // K = k0+bKoff .. +7
            b[u].q[1] = *(const uint4*)(p + 16);   // K = k0+bKoff+16 .. +23
        }
#pragma unroll
        for (int t = 0; t < 4; ++t)
#pragma unroll
            for (int u = 0; u < 2; ++u)
                acc[t][u] = __builtin_amdgcn_wmma_f32_16x16x32_bf16(
                    false, a[t].v, false, b[u].v,
                    (short)0, acc[t][u], false, false);
    }

    // C/D layout: lane<16 -> (M=r, N=lane); lane>=16 -> (M=8+r, N=lane-16)
#pragma unroll
    for (int t = 0; t < 4; ++t) {
#pragma unroll
        for (int u = 0; u < 2; ++u) {
            const int col = nb + u * 16 + l15;
            const float bv = bias[col];
#pragma unroll
            for (int r = 0; r < 8; ++r) {
                const int row = mb + t * 16 + lhi * 8 + r;
                C[(size_t)row * N + col] = acc[t][u][r] + bv;
            }
        }
    }
}

// ---------------------------------------------------------------------------
// vsum[b, e] = sum_s vproj[b, s, e]   (chunked, atomic accumulate)
// grid: (E/256, S/64 chunks, B), block 256
// ---------------------------------------------------------------------------
__global__ __launch_bounds__(256) void vsum_kernel(const float* __restrict__ vproj,
                                                   float* __restrict__ vsum) {
    const int e  = blockIdx.x * 256 + threadIdx.x;
    const int b  = blockIdx.z;
    const int s0 = blockIdx.y * 64;
    float acc = 0.f;
    for (int s = s0; s < s0 + 64; ++s)
        acc += vproj[((size_t)b * SS + s) * EE + e];
    atomicAdd(&vsum[b * EE + e], acc);
}

// ---------------------------------------------------------------------------
// Banded attention + residual. One wave per query row i.
//   s_j   = (q_i . k_j)/8  for |i-j|<=16, j in [0,S)
//   softmax_t over { s_j * am_j^(t+1) (in-band), 0 (all other S-N_in cols) }
//   context_i = sum_in (p_j - p0) v_j + p0 * Vsum      (per window, weighted)
//   x = q_i + context_i
// grid: (S/8, B*H), block 256 (= 8 waves)
// ---------------------------------------------------------------------------
__global__ __launch_bounds__(256) void band_attn_kernel(
    const float* __restrict__ qproj, const float* __restrict__ kproj,
    const float* __restrict__ vproj, const float* __restrict__ vsum,
    const float* __restrict__ amask, const float* __restrict__ wwin,
    float* __restrict__ xbuf)
{
    const int wave = threadIdx.x >> 5;
    const int lane = threadIdx.x & 31;
    const int i  = blockIdx.x * 8 + wave;
    const int bh = blockIdx.y;
    const int b  = bh / HH;
    const int h  = bh % HH;
    const size_t rowQ = ((size_t)b * SS + i) * EE + h * DD;

    const float qa = qproj[rowQ + lane];
    const float qb = qproj[rowQ + lane + 32];
    const float wt[3] = { wwin[0], wwin[1], wwin[2] };
    const float scale = 0.125f;  // 1/sqrt(64)

    float sc[BANDW];   // raw scaled scores (broadcast to all lanes)
    float amv[BANDW];  // mask value per key column
    int nIn = 0;
#pragma unroll
    for (int jj = 0; jj < BANDW; ++jj) {
        const int j = i - WIN + jj;
        const bool valid = (j >= 0) && (j < SS);
        float part = 0.f, am = 0.f;
        if (valid) {
            const size_t rowK = ((size_t)b * SS + j) * EE + h * DD;
            part = qa * kproj[rowK + lane] + qb * kproj[rowK + lane + 32];
            am = amask[b * SS + j];
        }
#pragma unroll
        for (int off = 16; off; off >>= 1)
            part += __shfl_xor(part, off, 32);
        sc[jj]  = part * scale;
        amv[jj] = am;
        nIn += valid ? 1 : 0;
    }

    float ctxw[BANDW];
    float amp[BANDW];
#pragma unroll
    for (int jj = 0; jj < BANDW; ++jj) { ctxw[jj] = 0.f; amp[jj] = 1.f; }
    float g = 0.f;

    for (int t = 0; t < 3; ++t) {
#pragma unroll
        for (int jj = 0; jj < BANDW; ++jj) amp[jj] *= amv[jj];  // am^(t+1)
        float m = 0.f;  // out-of-band zeros always participate => max >= 0
#pragma unroll
        for (int jj = 0; jj < BANDW; ++jj) {
            const int j = i - WIN + jj;
            if (j >= 0 && j < SS) m = fmaxf(m, sc[jj] * amp[jj]);
        }
        float Z = (float)(SS - nIn) * __expf(-m);
#pragma unroll
        for (int jj = 0; jj < BANDW; ++jj) {
            const int j = i - WIN + jj;
            if (j >= 0 && j < SS) Z += __expf(sc[jj] * amp[jj] - m);
        }
        const float inv = 1.0f / Z;
        const float p0  = __expf(-m) * inv;
        g += wt[t] * p0;
#pragma unroll
        for (int jj = 0; jj < BANDW; ++jj) {
            const int j = i - WIN + jj;
            if (j >= 0 && j < SS)
                ctxw[jj] += wt[t] * (__expf(sc[jj] * amp[jj] - m) * inv - p0);
        }
    }

    float ca = 0.f, cb = 0.f;
#pragma unroll
    for (int jj = 0; jj < BANDW; ++jj) {
        const int j = i - WIN + jj;
        if (j >= 0 && j < SS) {
            const size_t rowV = ((size_t)b * SS + j) * EE + h * DD;
            ca += ctxw[jj] * vproj[rowV + lane];
            cb += ctxw[jj] * vproj[rowV + lane + 32];
        }
    }
    const float* vs = vsum + (size_t)b * EE + h * DD;
    ca += g * vs[lane];
    cb += g * vs[lane + 32];

    xbuf[rowQ + lane]      = qa + ca;
    xbuf[rowQ + lane + 32] = qb + cb;
}

// ---------------------------------------------------------------------------
// LayerNorm over last dim (768). One block (256 threads) per row.
// ---------------------------------------------------------------------------
__global__ __launch_bounds__(256) void layernorm_kernel(
    const float* __restrict__ x, const float* __restrict__ gamma,
    const float* __restrict__ beta, float* __restrict__ out)
{
    __shared__ float r1[8], r2[8];
    const int row = blockIdx.x;
    const float* xr = x + (size_t)row * EE;
    float sum = 0.f, sumsq = 0.f;
    for (int e = threadIdx.x; e < EE; e += 256) {
        const float v = xr[e];
        sum += v; sumsq += v * v;
    }
#pragma unroll
    for (int off = 16; off; off >>= 1) {
        sum   += __shfl_xor(sum,   off, 32);
        sumsq += __shfl_xor(sumsq, off, 32);
    }
    const int wave = threadIdx.x >> 5, lane = threadIdx.x & 31;
    if (lane == 0) { r1[wave] = sum; r2[wave] = sumsq; }
    __syncthreads();
    if (wave == 0) {
        float a = (lane < 8) ? r1[lane] : 0.f;
        float c = (lane < 8) ? r2[lane] : 0.f;
#pragma unroll
        for (int off = 4; off; off >>= 1) {
            a += __shfl_xor(a, off, 32);
            c += __shfl_xor(c, off, 32);
        }
        if (lane == 0) { r1[0] = a; r2[0] = c; }
    }
    __syncthreads();
    const float mean = r1[0] * (1.0f / EE);
    const float var  = r2[0] * (1.0f / EE) - mean * mean;
    const float rstd = rsqrtf(var + 1e-5f);
    for (int e = threadIdx.x; e < EE; e += 256) {
        const float v = (xr[e] - mean) * rstd;
        out[(size_t)row * EE + e] = gamma[e] * v + beta[e];
    }
}

// ---------------------------------------------------------------------------
// Host-side orchestration
// ---------------------------------------------------------------------------
extern "C" void kernel_launch(void* const* d_in, const int* in_sizes, int n_in,
                              void* d_out, int out_size, void* d_ws, size_t ws_size,
                              hipStream_t stream) {
    (void)in_sizes; (void)n_in; (void)out_size; (void)ws_size;

    const float* q_in  = (const float*)d_in[0];
    const float* k_in  = (const float*)d_in[1];
    const float* v_in  = (const float*)d_in[2];
    const float* amask = (const float*)d_in[3];
    const float* Wq    = (const float*)d_in[4];
    const float* bq    = (const float*)d_in[5];
    const float* Wk    = (const float*)d_in[6];
    const float* bk    = (const float*)d_in[7];
    const float* Wv    = (const float*)d_in[8];
    const float* bv    = (const float*)d_in[9];
    const float* wwin  = (const float*)d_in[10];
    const float* gamma = (const float*)d_in[11];
    const float* beta  = (const float*)d_in[12];
    float* out = (float*)d_out;

    // Workspace carve-up (total ~73 MB)
    char* ws = (char*)d_ws;
    size_t off = 0;
    auto carve = [&](size_t bytes) -> void* {
        void* p = ws + off;
        off += (bytes + 255) & ~(size_t)255;
        return p;
    };
    unsigned short* qb16 = (unsigned short*)carve((size_t)MM * EE * 2);
    unsigned short* kb16 = (unsigned short*)carve((size_t)MM * EE * 2);
    unsigned short* vb16 = (unsigned short*)carve((size_t)MM * EE * 2);
    unsigned short* wq16 = (unsigned short*)carve((size_t)EE * EE * 2);
    unsigned short* wk16 = (unsigned short*)carve((size_t)EE * EE * 2);
    unsigned short* wv16 = (unsigned short*)carve((size_t)EE * EE * 2);
    float* qproj = (float*)carve((size_t)MM * EE * 4);
    float* kproj = (float*)carve((size_t)MM * EE * 4);
    float* vproj = (float*)carve((size_t)MM * EE * 4);
    float* xbuf  = (float*)carve((size_t)MM * EE * 4);
    float* vsum  = (float*)carve((size_t)BB * EE * 4);

    // 1) f32 -> bf16 conversions (activations + weights)
    {
        const int nAct = MM * EE;             // 3,145,728
        const int nW   = EE * EE;             // 589,824
        cvt_bf16_kernel<<<(nAct + 255) / 256, 256, 0, stream>>>(q_in, qb16, nAct);
        cvt_bf16_kernel<<<(nAct + 255) / 256, 256, 0, stream>>>(k_in, kb16, nAct);
        cvt_bf16_kernel<<<(nAct + 255) / 256, 256, 0, stream>>>(v_in, vb16, nAct);
        cvt_bf16_kernel<<<(nW + 255) / 256, 256, 0, stream>>>(Wq, wq16, nW);
        cvt_bf16_kernel<<<(nW + 255) / 256, 256, 0, stream>>>(Wk, wk16, nW);
        cvt_bf16_kernel<<<(nW + 255) / 256, 256, 0, stream>>>(Wv, wv16, nW);
    }

    // 2) Q/K/V projections: [4096,768] = [4096,768] x [768,768]^T + bias
    {
        dim3 grid(EE / 128, MM / 128);  // (6, 32)
        wmma_gemm_bias_kernel<<<grid, 256, 0, stream>>>(qb16, wq16, bq, qproj, MM, EE, EE);
        wmma_gemm_bias_kernel<<<grid, 256, 0, stream>>>(kb16, wk16, bk, kproj, MM, EE, EE);
        wmma_gemm_bias_kernel<<<grid, 256, 0, stream>>>(vb16, wv16, bv, vproj, MM, EE, EE);
    }

    // 3) Per-(b,e) column sum of V  (for the uniform out-of-band softmax mass)
    hipMemsetAsync(vsum, 0, (size_t)BB * EE * sizeof(float), stream);
    {
        dim3 grid(EE / 256, SS / 64, BB);  // (3, 16, 4)
        vsum_kernel<<<grid, 256, 0, stream>>>(vproj, vsum);
    }

    // 4) Banded attention (+ residual q)  -> xbuf
    {
        dim3 grid(SS / 8, BB * HH);  // (128, 48)
        band_attn_kernel<<<grid, 256, 0, stream>>>(qproj, kproj, vproj, vsum,
                                                   amask, wwin, xbuf);
    }

    // 5) LayerNorm -> d_out
    layernorm_kernel<<<MM, 256, 0, stream>>>(xbuf, gamma, beta, out);
}
